// IRadonMap_31928786878825
// MI455X (gfx1250) — matrix operand order repaired and checked by reading, not compiled, not run on hardware
//
#include <hip/hip_runtime.h>
#include <hip/hip_bf16.h>

// ---------------------------------------------------------------------------
// IRadonMap for MI455X (gfx1250, wave32, WMMA)
//   Stage 0: fp32 -> bf16 conversion of sinogram S [1440,512] and W [512,512]
//   Stage 1: WMMA bf16 GEMM  Xf[m,e] = sum_d S[m,d] * W[e,d]   (fp32 accum)
//            register-tiled 32(M) x 64(N) per wave: 8 WMMAs / K-step
//   Stage 2: gather-backprojection  out[b,pix] = sum_a Xf[b, cm[pix,a]]*wt
// ---------------------------------------------------------------------------

typedef __attribute__((ext_vector_type(16))) __bf16 v16bf;
typedef __attribute__((ext_vector_type(8)))  __bf16 v8bf;
typedef __attribute__((ext_vector_type(8)))  float  v8f;

#define BATCH      8
#define NUM_ANGLES 180
#define NUM_DET    512
#define M_TOT      (BATCH * NUM_ANGLES)   // 1440
#define K_TOT      NUM_DET                // 512
#define N_TOT      NUM_DET                // 512
#define NPIX       (256 * 256)            // 65536
#define PLANE      (NUM_ANGLES * NUM_DET) // 92160 elements per batch

// ---- Stage 0: fp32 -> bf16 (round to nearest even) ------------------------
__global__ void cvt_f32_to_bf16(const float* __restrict__ in,
                                unsigned short* __restrict__ out, int n) {
  int i = blockIdx.x * blockDim.x + threadIdx.x;
  if (i < n) {
    unsigned int u = __float_as_uint(in[i]);
    unsigned int r = u + 0x7FFFu + ((u >> 16) & 1u);   // RNE
    out[i] = (unsigned short)(r >> 16);
  }
}

// ---- Stage 1: bf16 WMMA GEMM, 32x64 register tile per wave ----------------
// A = S tile (rows of (b,a), K = detector d), row-major [M_TOT, K_TOT]
// B[k,n] = W[n,k]  -> column n of B is row n of W (contiguous)
__device__ __forceinline__ v16bf load_a_frag(const __bf16* arow, int k0, int khalf) {
  // 16-bit A 16x32 layout (contiguous, matches B):
  //   V0..V7: K = khalf*16 + 0..15
  v16bf a = *reinterpret_cast<const v16bf*>(arow + k0 + khalf * 16);     // 32B
  return a;
}

__global__ void wmma_filter_gemm(const unsigned short* __restrict__ Sb,
                                 const unsigned short* __restrict__ Wb,
                                 float* __restrict__ Xf) {
  const int lane  = threadIdx.x & 31;
  const int wave  = (blockIdx.x * blockDim.x + threadIdx.x) >> 5; // 0..359
  const int tn4   = wave & 7;           // N group: 64 columns
  const int tm2   = wave >> 3;          // M group: 32 rows (0..44, exact)

  const int mrow  = lane & 15;          // M (for A) / N (for B) within tile
  const int khalf = lane >> 4;          // 0 or 1

  const __bf16* arow0 = (const __bf16*)Sb + (tm2 * 32 + mrow) * K_TOT;
  const __bf16* arow1 = arow0 + 16 * K_TOT;
  const __bf16* brow  = (const __bf16*)Wb + (tn4 * 64 + mrow) * K_TOT;

  v8f acc[2][4];
  #pragma unroll
  for (int mi = 0; mi < 2; ++mi)
    #pragma unroll
    for (int j = 0; j < 4; ++j) acc[mi][j] = (v8f){};

  #pragma unroll 2
  for (int k0 = 0; k0 < K_TOT; k0 += 32) {
    const v16bf a0 = load_a_frag(arow0, k0, khalf);
    const v16bf a1 = load_a_frag(arow1, k0, khalf);
    #pragma unroll
    for (int j = 0; j < 4; ++j) {
      // B fragment (32x16): lane group khalf holds K = khalf*16 + 0..15
      const v16bf b =
          *reinterpret_cast<const v16bf*>(brow + j * 16 * K_TOT + k0 + khalf * 16);
      acc[0][j] = __builtin_amdgcn_wmma_f32_16x16x32_bf16(
          false, a0, false, b, (short)0, acc[0][j], false, false);
      acc[1][j] = __builtin_amdgcn_wmma_f32_16x16x32_bf16(
          false, a1, false, b, (short)0, acc[1][j], false, false);
    }
  }

  // D layout: VGPR r -> M = r + 8*khalf, lane&15 -> N
  #pragma unroll
  for (int mi = 0; mi < 2; ++mi) {
    #pragma unroll
    for (int j = 0; j < 4; ++j) {
      float* outbase = Xf + (tn4 * 64 + j * 16) + (lane & 15);
      #pragma unroll
      for (int r = 0; r < 8; ++r) {
        const int mg = tm2 * 32 + mi * 16 + r + 8 * khalf;
        outbase[mg * N_TOT] = acc[mi][j][r];
      }
    }
  }
}

// ---- Stage 2: weighted backprojection (gather over 180 angles) ------------
// One thread per pixel; amortize coord/weight loads over all 8 batches and
// keep 8 independent FMA chains in flight to hide L2 gather latency.
__global__ void backproject(const float* __restrict__ Xf,      // [1440,512]
                            const int*   __restrict__ coord,   // [NPIX,180]
                            const float* __restrict__ wts,     // [NPIX,180]
                            float*       __restrict__ out) {   // [8,NPIX]
  const int pix = blockIdx.x * blockDim.x + threadIdx.x;
  if (pix >= NPIX) return;

  const int*   cm = coord + pix * NUM_ANGLES;
  const float* wp = wts   + pix * NUM_ANGLES;

  // Pull the per-pixel coord/weight streams toward the WGP caches.
  __builtin_prefetch(cm, 0, 0);
  __builtin_prefetch(wp, 0, 0);

  float acc[BATCH];
  #pragma unroll
  for (int b = 0; b < BATCH; ++b) acc[b] = 0.0f;

  #pragma unroll 4
  for (int a = 0; a < NUM_ANGLES; ++a) {
    const int   idx = cm[a];          // index into per-batch A*D plane
    const float w   = wp[a];
    const float* p  = Xf + idx;
    #pragma unroll
    for (int b = 0; b < BATCH; ++b)
      acc[b] = fmaf(p[b * PLANE], w, acc[b]);
  }

  #pragma unroll
  for (int b = 0; b < BATCH; ++b) out[b * NPIX + pix] = acc[b];
}

// ---------------------------------------------------------------------------
extern "C" void kernel_launch(void* const* d_in, const int* in_sizes, int n_in,
                              void* d_out, int out_size, void* d_ws, size_t ws_size,
                              hipStream_t stream) {
  const float* sino  = (const float*)d_in[0];  // [8,1,180,512]
  const float* W     = (const float*)d_in[1];  // [512,512]
  const float* wts   = (const float*)d_in[2];  // [1,1,256,256,180]
  const int*   coord = (const int*)  d_in[3];  // [256,256,180]
  float*       out   = (float*)d_out;          // [8,1,256,256]

  // Workspace layout (≈4.7 MB): Xf fp32 | S bf16 | W bf16
  char* ws = (char*)d_ws;
  float*          Xf = (float*)ws;                                  // 2,949,120 B
  unsigned short* Sb = (unsigned short*)(ws + (size_t)M_TOT * N_TOT * 4);
  unsigned short* Wb = Sb + (size_t)M_TOT * K_TOT;                  // +1,474,560 B

  // Stage 0: conversions
  {
    const int nS = M_TOT * K_TOT;     // 737280
    cvt_f32_to_bf16<<<(nS + 255) / 256, 256, 0, stream>>>(sino, Sb, nS);
    const int nW = N_TOT * K_TOT;     // 262144
    cvt_f32_to_bf16<<<(nW + 255) / 256, 256, 0, stream>>>(W, Wb, nW);
  }

  // Stage 1: WMMA GEMM. 45 x 8 = 360 wave-tiles (32x64 each), 8 waves/block
  wmma_filter_gemm<<<45, 256, 0, stream>>>(Sb, Wb, Xf);

  // Stage 2: backprojection, one thread per pixel
  backproject<<<NPIX / 256, 256, 0, stream>>>(Xf, coord, wts, out);
}
